// NCM_classifier_86431921865191
// MI455X (gfx1250) — compile-verified, matrix operand order checked by reading, not run on hardware
//
#include <hip/hip_runtime.h>

// NCM classifier scores: out[n,c] = -sqrt(max(||x_n||^2 + ||m_c||^2 - 2 x_n.m_c, 0))
// N=16384 rows (x), C=2048 clusters (means), F=512 features. fp32 in/out.
//
// MI455X plan:
//  - Cross term = 16384x2048x512 GEMM via V_WMMA_F32_16X16X32_BF16 with
//    split-bf16 (hi+lo) fp32 emulation: acc += Ahi*Bhi + Ahi*Blo + Alo*Bhi.
//  - fp32->bf16 hi/lo split is done ONCE in a prepass into d_ws (x scaled by -2),
//    so the GEMM inner loop is pure b128 loads + WMMA (no per-iter VALU splitting).
//  - Inputs (36MB incl. split copies) fit in 192MB L2 -> no LDS staging needed;
//    fragment loads are coalesced b128s, redundancy across waves hits L2.
//  - 256-thread blocks (8 wave32), 128x128 block tile, 32x64 wave tile
//    (2x4 accumulators of 16x16), K loop: 16 steps of 32.

#define N_ROWS 16384
#define N_CLUST 2048
#define N_FEAT 512

typedef __attribute__((ext_vector_type(16))) __bf16 v16bf;
typedef __attribute__((ext_vector_type(8)))  __bf16 v8bf;
typedef __attribute__((ext_vector_type(4)))  __bf16 v4bf;
typedef __attribute__((ext_vector_type(8)))  float  v8f;
typedef __attribute__((ext_vector_type(4)))  float  f32x4;

// Split fp32 into bf16 hi (truncated) + bf16 lo (rounded residual): x ~= hi + lo
__device__ __forceinline__ void split_bf16(float x, __bf16& hi, __bf16& lo) {
    unsigned u  = __builtin_bit_cast(unsigned, x);
    unsigned hu = u & 0xFFFF0000u;
    float    hf = __builtin_bit_cast(float, hu);
    hi = __builtin_bit_cast(__bf16, (unsigned short)(hu >> 16));
    float r  = x - hf;                                  // |r| < 2^-7 |x|
    unsigned ru = __builtin_bit_cast(unsigned, r);
    unsigned rr = ru + 0x7FFFu + ((ru >> 16) & 1u);     // RNE to bf16
    lo = __builtin_bit_cast(__bf16, (unsigned short)(rr >> 16));
}

// Prepass: elementwise fp32 -> (hi, lo) bf16 arrays, with scale folded in.
// 4 elements per thread, b128 in / b64 out, fully coalesced.
__global__ __launch_bounds__(256) void ncm_split_kernel(
        const float* __restrict__ src, __bf16* __restrict__ hi,
        __bf16* __restrict__ lo, float scale, int count4) {
    int t = blockIdx.x * blockDim.x + threadIdx.x;
    if (t >= count4) return;
    int idx = t * 4;
    f32x4 f = *(const f32x4*)(src + idx);
    v4bf h, l;
#pragma unroll
    for (int i = 0; i < 4; ++i) {
        __bf16 hh, ll;
        split_bf16(f[i] * scale, hh, ll);
        h[i] = hh;
        l[i] = ll;
    }
    *(v4bf*)(hi + idx) = h;
    *(v4bf*)(lo + idx) = l;
}

// Row sum-of-squares: one wave32 per row of 512 floats.
__global__ __launch_bounds__(256) void ncm_rowsq(const float* __restrict__ src,
                                                 float* __restrict__ dst,
                                                 int nrows) {
    int lane = threadIdx.x & 31;
    int row  = blockIdx.x * 8 + (threadIdx.x >> 5);
    if (row >= nrows) return;
    const float* p = src + (size_t)row * N_FEAT;
    float s = 0.0f;
#pragma unroll 4
    for (int i = lane; i < N_FEAT; i += 32) {
        float v = p[i];
        s += v * v;
    }
#pragma unroll
    for (int off = 16; off > 0; off >>= 1)
        s += __shfl_xor(s, off, 32);
    if (lane == 0) dst[row] = s;
}

// Per-lane bf16 fragment load, 16x32 tile of a row-major bf16 matrix (stride F).
// ISA 16-bit A 16x32 layout: lane r=lane&15 is row; kb=(lane>>4)*8 selects K
// group. Elements [0..7] = K cb..cb+7, elements [8..15] = K cb+16..cb+23.
// B (32x16) mirrors with N in place of M -> same loader on the means arrays.
__device__ __forceinline__ v16bf load_bf_frag(const __bf16* __restrict__ rowptr,
                                              int cb) {
    v8bf p0 = *(const v8bf*)(rowptr + cb);        // 16B -> global_load_b128
    v8bf p1 = *(const v8bf*)(rowptr + cb + 16);   // 16B -> global_load_b128
    return __builtin_shufflevector(p0, p1, 0, 1, 2, 3, 4, 5, 6, 7,
                                   8, 9, 10, 11, 12, 13, 14, 15);
}

__global__ __launch_bounds__(256) void ncm_gemm(
        const __bf16* __restrict__ xhi, const __bf16* __restrict__ xlo,
        const __bf16* __restrict__ mhi, const __bf16* __restrict__ mlo,
        const float* __restrict__ xsq, const float* __restrict__ msq,
        float* __restrict__ out) {
    const int lane = threadIdx.x & 31;
    const int wave = threadIdx.x >> 5;
    const int wm   = wave & 3;   // 4 row slices of 32
    const int wn   = wave >> 2;  // 2 col slices of 64
    const int blkM = blockIdx.y * 128;  // x rows
    const int blkC = blockIdx.x * 128;  // mean rows (output cols)
    const int r    = lane & 15;
    const int kb8  = (lane >> 4) * 8;

    const int mrow0 = blkM + wm * 32;
    const int ccol0 = blkC + wn * 64;

    v8f acc[2][4];
#pragma unroll
    for (int mi = 0; mi < 2; ++mi)
#pragma unroll
        for (int ni = 0; ni < 4; ++ni)
#pragma unroll
            for (int v = 0; v < 8; ++v)
                acc[mi][ni][v] = 0.0f;

    const __bf16* xh[2];
    const __bf16* xl[2];
#pragma unroll
    for (int mi = 0; mi < 2; ++mi) {
        size_t off = (size_t)(mrow0 + mi * 16 + r) * N_FEAT;
        xh[mi] = xhi + off;
        xl[mi] = xlo + off;
    }
    const __bf16* ch[4];
    const __bf16* cl[4];
#pragma unroll
    for (int ni = 0; ni < 4; ++ni) {
        size_t off = (size_t)(ccol0 + ni * 16 + r) * N_FEAT;
        ch[ni] = mhi + off;
        cl[ni] = mlo + off;
    }

    for (int kk = 0; kk < N_FEAT; kk += 32) {
        const int cb = kk + kb8;
        v16bf ahi[2], alo[2];
#pragma unroll
        for (int mi = 0; mi < 2; ++mi) {
            ahi[mi] = load_bf_frag(xh[mi], cb);
            alo[mi] = load_bf_frag(xl[mi], cb);
        }
#pragma unroll
        for (int ni = 0; ni < 4; ++ni) {
            v16bf bhi = load_bf_frag(ch[ni], cb);
            v16bf blo = load_bf_frag(cl[ni], cb);
#pragma unroll
            for (int mi = 0; mi < 2; ++mi) {
                acc[mi][ni] = __builtin_amdgcn_wmma_f32_16x16x32_bf16(
                    false, ahi[mi], false, bhi, (short)0, acc[mi][ni], false, false);
                acc[mi][ni] = __builtin_amdgcn_wmma_f32_16x16x32_bf16(
                    false, ahi[mi], false, blo, (short)0, acc[mi][ni], false, false);
                acc[mi][ni] = __builtin_amdgcn_wmma_f32_16x16x32_bf16(
                    false, alo[mi], false, bhi, (short)0, acc[mi][ni], false, false);
            }
        }
    }

    // Epilogue: d2 = ||x||^2 + ||m||^2 - 2 x.m (acc), out = -sqrt(max(d2,0)).
    // C/D layout: lane holds N=r; VGPR v holds M = kb8 + v within the 16x16 tile.
#pragma unroll
    for (int mi = 0; mi < 2; ++mi) {
        float xs[8];
#pragma unroll
        for (int v = 0; v < 8; ++v)
            xs[v] = xsq[mrow0 + mi * 16 + kb8 + v];
#pragma unroll
        for (int ni = 0; ni < 4; ++ni) {
            const int n  = ccol0 + ni * 16 + r;
            const float ms = msq[n];
#pragma unroll
            for (int v = 0; v < 8; ++v) {
                const int m = mrow0 + mi * 16 + kb8 + v;
                float d2 = xs[v] + ms + acc[mi][ni][v];
                d2 = fmaxf(d2, 0.0f);
                out[(size_t)m * N_CLUST + n] = -sqrtf(d2);
            }
        }
    }
}

extern "C" void kernel_launch(void* const* d_in, const int* in_sizes, int n_in,
                              void* d_out, int out_size, void* d_ws, size_t ws_size,
                              hipStream_t stream) {
    const float* x     = (const float*)d_in[0];
    const float* means = (const float*)d_in[1];
    float* out = (float*)d_out;

    // Workspace layout (~36.1 MB):
    //   xhi [N*F bf16] | xlo [N*F bf16] | mhi [C*F bf16] | mlo [C*F bf16]
    //   | xsq [N f32] | msq [C f32]
    __bf16* xhi = (__bf16*)d_ws;
    __bf16* xlo = xhi + (size_t)N_ROWS * N_FEAT;
    __bf16* mhi = xlo + (size_t)N_ROWS * N_FEAT;
    __bf16* mlo = mhi + (size_t)N_CLUST * N_FEAT;
    float*  xsq = (float*)(mlo + (size_t)N_CLUST * N_FEAT);
    float*  msq = xsq + N_ROWS;

    const int xcount4 = N_ROWS * N_FEAT / 4;   // 2,097,152
    const int mcount4 = N_CLUST * N_FEAT / 4;  //   262,144
    ncm_split_kernel<<<xcount4 / 256, 256, 0, stream>>>(x, xhi, xlo, -2.0f, xcount4);
    ncm_split_kernel<<<mcount4 / 256, 256, 0, stream>>>(means, mhi, mlo, 1.0f, mcount4);
    ncm_rowsq<<<N_ROWS / 8, 256, 0, stream>>>(x, xsq, N_ROWS);
    ncm_rowsq<<<N_CLUST / 8, 256, 0, stream>>>(means, msq, N_CLUST);

    dim3 grid(N_CLUST / 128, N_ROWS / 128);  // (16, 128)
    ncm_gemm<<<grid, 256, 0, stream>>>(xhi, xlo, mhi, mlo, xsq, msq, out);
}
